// RNAProteinEncoderLayer_86165633892437
// MI455X (gfx1250) — matrix operand order repaired and checked by reading, not compiled, not run on hardware
//
#include <hip/hip_runtime.h>
#include <hip/hip_bf16.h>

#define D_    512
#define H_    8
#define HD_   64
#define DFF_  2048
#define S1_   1024
#define S2_   512
#define B_    8
#define NT1   (S1_ * B_)   /* 8192 rna tokens     */
#define NT2   (S2_ * B_)   /* 4096 protein tokens */

typedef __attribute__((ext_vector_type(16))) _Float16      v16h;
typedef __attribute__((ext_vector_type(8)))  float         v8f;
typedef __attribute__((ext_vector_type(4)))  unsigned int  v4u;
typedef __attribute__((ext_vector_type(8)))  int           v8i32;
typedef __attribute__((ext_vector_type(4)))  int           v4i32;

// ---------------------------------------------------------------------------
// WMMA fragment helpers (gfx1250 wave32, 16x16x32 f16; this file's own contiguous-k order, same for A and B)
// A-frag: lane row = lane&15, halves = 16 contiguous K starting at (lane>>4)*16
// B-frag: lane col = lane&15, halves = 16 contiguous K starting at (lane>>4)*16
// ---------------------------------------------------------------------------
__device__ __forceinline__ v16h frag_a(const _Float16* p) {
  union { v16h v; uint4 q[2]; } u;
  u.q[0] = *reinterpret_cast<const uint4*>(p);
  u.q[1] = *reinterpret_cast<const uint4*>(p + 8);
  return u.v;
}
__device__ __forceinline__ v16h frag_b(const _Float16* p) {
  union { v16h v; uint4 q[2]; } u;
  u.q[0] = *reinterpret_cast<const uint4*>(p);
  u.q[1] = *reinterpret_cast<const uint4*>(p + 8);
  return u.v;
}
template <int STRIDE>  // compile-time element stride -> immediate offsets
__device__ __forceinline__ v16h frag_strided(const _Float16* p) {
  v16h f;
#pragma unroll
  for (int i = 0; i < 16; ++i) f[i] = p[(size_t)STRIDE * i];
  return f;
}
__device__ __forceinline__ v8f wmma16(v16h a, v16h b, v8f c) {
  return __builtin_amdgcn_wmma_f32_16x16x32_f16(false, a, false, b,
                                                (short)0, c, false, false);
}

// ---------------------------------------------------------------------------
// Tensor Data Mover: 2-D f16 tile load (tile_dim0 = k extent along contiguous
// axis, tile_dim1 = rows) from global into LDS at byte offset lds_off.
// D# packing per CDNA5 ISA ch.8 (group0 / group1; groups 2-3 zero => 2-D).
// This toolchain exposes the 6-arg builtin: (g0, g1, g2, g3, g4, cpol).
// ---------------------------------------------------------------------------
#if __has_builtin(__builtin_amdgcn_tensor_load_to_lds)
#define HAVE_TDM 1
__device__ __forceinline__ void tdm_load_tile_f16(unsigned lds_off,
                                                  const _Float16* gptr,
                                                  unsigned kelems, unsigned rows,
                                                  unsigned stride_elems) {
  unsigned long long ga = (unsigned long long)(uintptr_t)gptr;
  v4u g0;
  g0[0] = 1u;                                    // count=1, user mode
  g0[1] = lds_off;                               // lds_addr (bytes)
  g0[2] = (unsigned)(ga & 0xffffffffu);          // global_addr[31:0]
  g0[3] = (unsigned)((ga >> 32) & 0x01ffffffu)   // global_addr[56:32]
        | 0x80000000u;                           // type=2 ("image")
  v8i32 g1;
  g1[0] = (int)(1u << 16);                       // data_size=1 -> 2 bytes
  g1[1] = (int)((kelems & 0xffffu) << 16);       // tensor_dim0[15:0] @bit48
  g1[2] = (int)(((kelems >> 16) & 0xffffu) |     // tensor_dim0[31:16]
                ((rows & 0xffffu) << 16));       // tensor_dim1[15:0] @bit80
  g1[3] = (int)(((rows >> 16) & 0xffffu) |       // tensor_dim1[31:16]
                ((kelems & 0xffffu) << 16));     // tile_dim0 @bit112
  g1[4] = (int)(rows & 0xffffu);                 // tile_dim1 (tile_dim2=0)
  g1[5] = (int)(stride_elems);                   // tensor_dim0_stride[31:0]
  g1[6] = 0;                                     // stride hi + dim1_stride lo
  g1[7] = 0;
  v4i32 z4; z4[0] = 0; z4[1] = 0; z4[2] = 0; z4[3] = 0;
  v8i32 z8; z8[0] = 0; z8[1] = 0; z8[2] = 0; z8[3] = 0;
  z8[4] = 0; z8[5] = 0; z8[6] = 0; z8[7] = 0;
  __builtin_amdgcn_tensor_load_to_lds(g0, g1, z4, z4, z8, 0);
}
#else
#define HAVE_TDM 0
#endif

// ---------------------------------------------------------------------------
// fp32 -> fp16 weight conversion
// ---------------------------------------------------------------------------
__global__ __launch_bounds__(256)
void cvt_f32_f16(const float* __restrict__ s, _Float16* __restrict__ d, int n) {
  int i = blockIdx.x * 256 + threadIdx.x;
  if (i < n) d[i] = (_Float16)s[i];
}

// ---------------------------------------------------------------------------
// LayerNorm: one wave per token (D=512 -> 16 elements/lane), fp32 in, f16 out
// ---------------------------------------------------------------------------
__global__ __launch_bounds__(256)
void layernorm_f16(const float* __restrict__ x, const float* __restrict__ g,
                   const float* __restrict__ be, _Float16* __restrict__ y,
                   int ntok) {
  const int lane = threadIdx.x & 31;
  const int wid  = threadIdx.x >> 5;
  const int tok  = blockIdx.x * 8 + wid;
  if (tok >= ntok) return;
  const float4* xp = reinterpret_cast<const float4*>(
      x + (size_t)tok * D_ + lane * 16);
  float v[16];
  float4 t;
#pragma unroll
  for (int c = 0; c < 4; ++c) {
    t = xp[c];
    v[c * 4 + 0] = t.x; v[c * 4 + 1] = t.y; v[c * 4 + 2] = t.z; v[c * 4 + 3] = t.w;
  }
  float s = 0.f;
#pragma unroll
  for (int i = 0; i < 16; ++i) s += v[i];
  s += __shfl_xor(s, 1, 32);  s += __shfl_xor(s, 2, 32);
  s += __shfl_xor(s, 4, 32);  s += __shfl_xor(s, 8, 32);
  s += __shfl_xor(s, 16, 32);
  const float mean = s * (1.f / D_);
  float vs = 0.f;
#pragma unroll
  for (int i = 0; i < 16; ++i) { float d = v[i] - mean; vs += d * d; }
  vs += __shfl_xor(vs, 1, 32);  vs += __shfl_xor(vs, 2, 32);
  vs += __shfl_xor(vs, 4, 32);  vs += __shfl_xor(vs, 8, 32);
  vs += __shfl_xor(vs, 16, 32);
  const float rstd = rsqrtf(vs * (1.f / D_) + 1e-5f);
  const int d0 = lane * 16;
  _Float16* yp = y + (size_t)tok * D_ + d0;
#pragma unroll
  for (int i = 0; i < 16; ++i)
    yp[i] = (_Float16)((v[i] - mean) * rstd * g[d0 + i] + be[d0 + i]);
}

// ---------------------------------------------------------------------------
// GEMM: Y[M,N] = X[M,K] @ W[N,K]^T + bias, WMMA f16->f32.
// All leading dims + K are template constants -> immediate-offset addressing.
// Wave tile 16(M) x 64(N); block = 8 waves = 128 x 64. Grid (N/64, M/128).
// The shared 64x32 weight tile is staged in LDS, double-buffered: the Tensor
// Data Mover (TENSORcnt) DMAs the next k-slice while waves compute from the
// current one. Dynamic LDS = 2 * 4096 B, descriptor lds_addr = 0 / 4096.
// ---------------------------------------------------------------------------
template <int WRITE_HALF, int RELU, int ADD_RES, int LDX, int LDW, int LDY, int KK>
__global__ __launch_bounds__(256)
void gemm_wmma(const _Float16* __restrict__ X,
               const _Float16* __restrict__ W,
               const float* __restrict__ bias,
               _Float16* __restrict__ Yh, float* __restrict__ Yf) {
  extern __shared__ _Float16 ldsB[];           // [2][64*32]
  const int lane = threadIdx.x & 31;
  const int wid  = threadIdx.x >> 5;
  const int m0   = blockIdx.y * 128 + wid * 16;
  const int n0   = blockIdx.x * 64;
  const int l15  = lane & 15;
  const int kba  = (lane >> 4) * 16;
  const int kbb  = (lane >> 4) * 16;
  const _Float16* pa = X + (size_t)(m0 + l15) * LDX + kba;
  const _Float16* pw = W + (size_t)n0 * LDW;   // 64-row weight panel
  v8f c0 = {}, c1 = {}, c2 = {}, c3 = {};

#if HAVE_TDM
  if (wid == 0) {
    tdm_load_tile_f16(0u, pw, 32u, 64u, (unsigned)LDW);
    __builtin_amdgcn_s_wait_tensorcnt(0);
  }
  __syncthreads();
#else
  {
    const int row = wid * 8 + (lane >> 2);
    const int col = (lane & 3) * 8;
    *reinterpret_cast<uint4*>(&ldsB[row * 32 + col]) =
        *reinterpret_cast<const uint4*>(pw + (size_t)row * LDW + col);
  }
  __syncthreads();
#endif

  int buf = 0;
  for (int k0 = 0; k0 < KK; k0 += 32) {
    const int has_next = (k0 + 32 < KK);
#if HAVE_TDM
    if (wid == 0 && has_next)
      tdm_load_tile_f16((unsigned)((buf ^ 1) * 4096),
                        pw + k0 + 32, 32u, 64u, (unsigned)LDW);
#endif
    const _Float16* bt = ldsB + buf * (64 * 32) + kbb;
    v16h a  = frag_a(pa + k0);
    __builtin_prefetch(pa + k0 + 128, 0, 1);
    v16h b0 = frag_b(bt + (l15) * 32);
    v16h b1 = frag_b(bt + (16 + l15) * 32);
    v16h b2 = frag_b(bt + (32 + l15) * 32);
    v16h b3 = frag_b(bt + (48 + l15) * 32);
    c0 = wmma16(a, b0, c0);
    c1 = wmma16(a, b1, c1);
    c2 = wmma16(a, b2, c2);
    c3 = wmma16(a, b3, c3);
#if HAVE_TDM
    if (wid == 0 && has_next) __builtin_amdgcn_s_wait_tensorcnt(0);
    __syncthreads();
    buf ^= 1;
#else
    __syncthreads();
    if (has_next) {
      const int row = wid * 8 + (lane >> 2);
      const int col = (lane & 3) * 8;
      *reinterpret_cast<uint4*>(&ldsB[(buf ^ 1) * (64 * 32) + row * 32 + col]) =
          *reinterpret_cast<const uint4*>(pw + (size_t)row * LDW + k0 + 32 + col);
    }
    __syncthreads();
    buf ^= 1;
#endif
  }

  const int   mrow = m0 + (lane >> 4) * 8;
  const float bb0 = bias[n0 + l15];
  const float bb1 = bias[n0 + 16 + l15];
  const float bb2 = bias[n0 + 32 + l15];
  const float bb3 = bias[n0 + 48 + l15];
#pragma unroll
  for (int r = 0; r < 8; ++r) {
    const size_t row = (size_t)(mrow + r) * LDY + n0 + l15;
    float v0 = c0[r] + bb0, v1 = c1[r] + bb1, v2 = c2[r] + bb2, v3 = c3[r] + bb3;
    if (RELU) {
      v0 = fmaxf(v0, 0.f); v1 = fmaxf(v1, 0.f);
      v2 = fmaxf(v2, 0.f); v3 = fmaxf(v3, 0.f);
    }
    if (WRITE_HALF) {
      Yh[row]      = (_Float16)v0;
      Yh[row + 16] = (_Float16)v1;
      Yh[row + 32] = (_Float16)v2;
      Yh[row + 48] = (_Float16)v3;
    } else if (ADD_RES) {
      Yf[row]      += v0;
      Yf[row + 16] += v1;
      Yf[row + 32] += v2;
      Yf[row + 48] += v3;
    } else {
      Yf[row]      = v0;
      Yf[row + 16] = v1;
      Yf[row + 32] = v2;
      Yf[row + 48] = v3;
    }
  }
}

// ---------------------------------------------------------------------------
// Flash attention: one wave per (b, h, 16-query tile), 4 waves/block.
// Tokens laid out (s, b): row index = s*B + b. hd = 64. Leading dims are
// template constants so the 16 strided V loads use immediate offsets.
// Online softmax over 32-key chunks; P goes through a per-wave LDS tile to be
// re-laid out from C-layout to A-layout for the P@V WMMA.
// ---------------------------------------------------------------------------
template <int LDQ, int LDK, int LDV>
__global__ __launch_bounds__(128)
void flash_attn(const _Float16* __restrict__ Q,
                const _Float16* __restrict__ Kp,
                const _Float16* __restrict__ Vp,
                _Float16* __restrict__ O,
                int Skv, float scale) {
  __shared__ _Float16 plds[4][16 * 32];
  const int lane = threadIdx.x & 31;
  const int wid  = threadIdx.x >> 5;
  const int l15  = lane & 15;
  const int hi   = lane >> 4;
  const int kba  = hi * 16;
  const int kbb  = hi * 16;
  const int q0   = (blockIdx.x * 4 + wid) * 16;
  const int b    = blockIdx.y >> 3;
  const int h    = blockIdx.y & 7;
  _Float16* pl   = plds[wid];

  const _Float16* qp = Q + ((size_t)(q0 + l15) * B_ + b) * LDQ + h * HD_ + kba;
  v16h aq0 = frag_a(qp);
  v16h aq1 = frag_a(qp + 32);

  constexpr int KSTEP = B_ * LDK;          // elements per key index
  constexpr int VSTEP = B_ * LDV;
  const _Float16* kp0 = Kp + ((size_t)l15 * B_ + b) * LDK + h * HD_ + kbb;
  const _Float16* vp0 = Vp + ((size_t)kbb * B_ + b) * LDV + h * HD_ + l15;

  v8f o0 = {}, o1 = {}, o2 = {}, o3 = {};
  float rmax[8], rsum[8];
#pragma unroll
  for (int r = 0; r < 8; ++r) { rmax[r] = -1e30f; rsum[r] = 0.f; }

  for (int kk = 0; kk < Skv; kk += 32) {
    // ---- scores: two 16x16 tiles (keys kk..kk+15, kk+16..kk+31) -----------
    v8f s0 = {}, s1 = {};
    {
      v16h bk0 = frag_b(kp0);
      v16h bk1 = frag_b(kp0 + 32);
      s0 = wmma16(aq0, bk0, s0);
      s0 = wmma16(aq1, bk1, s0);
      bk0 = frag_b(kp0 + 16 * KSTEP);
      bk1 = frag_b(kp0 + 16 * KSTEP + 32);
      s1 = wmma16(aq0, bk0, s1);
      s1 = wmma16(aq1, bk1, s1);
    }
    // ---- online softmax per accumulator row -------------------------------
#pragma unroll
    for (int r = 0; r < 8; ++r) {
      float v0 = s0[r] * scale, v1 = s1[r] * scale;
      float mx = fmaxf(v0, v1);
      mx = fmaxf(mx, __shfl_xor(mx, 1, 32));
      mx = fmaxf(mx, __shfl_xor(mx, 2, 32));
      mx = fmaxf(mx, __shfl_xor(mx, 4, 32));
      mx = fmaxf(mx, __shfl_xor(mx, 8, 32));
      const float mnew = fmaxf(rmax[r], mx);
      const float cf   = __expf(rmax[r] - mnew);
      rmax[r] = mnew;
      const float p0 = __expf(v0 - mnew);
      const float p1 = __expf(v1 - mnew);
      float ps = p0 + p1;
      ps += __shfl_xor(ps, 1, 32);
      ps += __shfl_xor(ps, 2, 32);
      ps += __shfl_xor(ps, 4, 32);
      ps += __shfl_xor(ps, 8, 32);
      rsum[r] = rsum[r] * cf + ps;
      o0[r] *= cf; o1[r] *= cf; o2[r] *= cf; o3[r] *= cf;
      const int m = hi * 8 + r;
      pl[m * 32 + l15]      = (_Float16)p0;
      pl[m * 32 + 16 + l15] = (_Float16)p1;
    }
    asm volatile("s_wait_dscnt 0x0" ::: "memory");
    // ---- P @ V: P as A-frag from LDS, V columns as strided B-frags --------
    v16h ap = frag_a(pl + l15 * 32 + kba);
    v16h bv;
    bv = frag_strided<VSTEP>(vp0);       o0 = wmma16(ap, bv, o0);
    bv = frag_strided<VSTEP>(vp0 + 16);  o1 = wmma16(ap, bv, o1);
    bv = frag_strided<VSTEP>(vp0 + 32);  o2 = wmma16(ap, bv, o2);
    bv = frag_strided<VSTEP>(vp0 + 48);  o3 = wmma16(ap, bv, o3);
    kp0 += 32 * (size_t)KSTEP;
    vp0 += 32 * (size_t)VSTEP;
  }
  // ---- normalize + store f16 ---------------------------------------------
#pragma unroll
  for (int r = 0; r < 8; ++r) {
    const float inv = 1.f / rsum[r];
    const int   s   = q0 + hi * 8 + r;
    _Float16* op = O + ((size_t)s * B_ + b) * D_ + h * HD_ + l15;
    op[0]  = (_Float16)(o0[r] * inv);
    op[16] = (_Float16)(o1[r] * inv);
    op[32] = (_Float16)(o2[r] * inv);
    op[48] = (_Float16)(o3[r] * inv);
  }
}

// ---------------------------------------------------------------------------
extern "C" void kernel_launch(void* const* d_in, const int* in_sizes, int n_in,
                              void* d_out, int out_size, void* d_ws,
                              size_t ws_size, hipStream_t stream) {
  (void)in_sizes; (void)n_in; (void)out_size; (void)ws_size;
  const float* rna    = (const float*)d_in[0];
  const float* prot   = (const float*)d_in[1];
  const float* Wqkv_s = (const float*)d_in[2];
  const float* bqkv_s = (const float*)d_in[3];
  const float* Wo_s   = (const float*)d_in[4];
  const float* bo_s   = (const float*)d_in[5];
  const float* Wqkv_c = (const float*)d_in[6];
  const float* bqkv_c = (const float*)d_in[7];
  const float* Wo_c   = (const float*)d_in[8];
  const float* bo_c   = (const float*)d_in[9];
  const float* W1     = (const float*)d_in[10];
  const float* b1     = (const float*)d_in[11];
  const float* W2     = (const float*)d_in[12];
  const float* b2     = (const float*)d_in[13];
  const float* g0     = (const float*)d_in[14];
  const float* be0    = (const float*)d_in[15];
  const float* g11    = (const float*)d_in[16];
  const float* be11   = (const float*)d_in[17];
  const float* g12    = (const float*)d_in[18];
  const float* be12   = (const float*)d_in[19];
  const float* g2     = (const float*)d_in[20];
  const float* be2    = (const float*)d_in[21];

  _Float16* ws = (_Float16*)d_ws;
  size_t off = 0;
  _Float16* wqkv_s_h = ws + off; off += (size_t)3 * D_ * D_;
  _Float16* wo_s_h   = ws + off; off += (size_t)D_ * D_;
  _Float16* wqkv_c_h = ws + off; off += (size_t)3 * D_ * D_;
  _Float16* wo_c_h   = ws + off; off += (size_t)D_ * D_;
  _Float16* w1_h     = ws + off; off += (size_t)DFF_ * D_;
  _Float16* w2_h     = ws + off; off += (size_t)D_ * DFF_;
  _Float16* hln      = ws + off; off += (size_t)NT1 * D_;
  _Float16* pln      = ws + off; off += (size_t)NT2 * D_;
  _Float16* qkv      = ws + off; off += (size_t)NT1 * 3 * D_;
  _Float16* attn     = ws + off; off += (size_t)NT1 * D_;
  _Float16* ffnmid   = ws + off; off += (size_t)NT1 * DFF_;

  float* x = (float*)d_out;  // residual stream lives in d_out
  (void)hipMemcpyAsync(x, rna, (size_t)NT1 * D_ * sizeof(float),
                       hipMemcpyDeviceToDevice, stream);

  cvt_f32_f16<<<(3 * D_ * D_) / 256, 256, 0, stream>>>(Wqkv_s, wqkv_s_h, 3 * D_ * D_);
  cvt_f32_f16<<<(D_ * D_) / 256, 256, 0, stream>>>(Wo_s, wo_s_h, D_ * D_);
  cvt_f32_f16<<<(3 * D_ * D_) / 256, 256, 0, stream>>>(Wqkv_c, wqkv_c_h, 3 * D_ * D_);
  cvt_f32_f16<<<(D_ * D_) / 256, 256, 0, stream>>>(Wo_c, wo_c_h, D_ * D_);
  cvt_f32_f16<<<(DFF_ * D_) / 256, 256, 0, stream>>>(W1, w1_h, DFF_ * D_);
  cvt_f32_f16<<<(D_ * DFF_) / 256, 256, 0, stream>>>(W2, w2_h, D_ * DFF_);

  const float  scale    = 0.125f;  // 1/sqrt(64)
  const size_t ldsBytes = 2 * 64 * 32 * sizeof(_Float16);  // 8 KB double-buffer

  // ---- 1) pre-norm self-attention ----------------------------------------
  layernorm_f16<<<NT1 / 8, 256, 0, stream>>>(x, g0, be0, hln, NT1);
  gemm_wmma<1, 0, 0, D_, D_, 3 * D_, D_>
      <<<dim3((3 * D_) / 64, NT1 / 128), 256, ldsBytes, stream>>>(
          hln, wqkv_s_h, bqkv_s, qkv, nullptr);
  flash_attn<3 * D_, 3 * D_, 3 * D_>
      <<<dim3(S1_ / 64, B_ * H_), 128, 0, stream>>>(
          qkv, qkv + D_, qkv + 2 * D_, attn, S1_, scale);
  gemm_wmma<0, 0, 1, D_, D_, D_, D_>
      <<<dim3(D_ / 64, NT1 / 128), 256, ldsBytes, stream>>>(
          attn, wo_s_h, bo_s, nullptr, x);

  // ---- 2) pre-norm cross-attention ---------------------------------------
  layernorm_f16<<<NT1 / 8, 256, 0, stream>>>(x, g11, be11, hln, NT1);
  layernorm_f16<<<NT2 / 8, 256, 0, stream>>>(prot, g12, be12, pln, NT2);
  _Float16* qc  = qkv;
  _Float16* kvc = qkv + (size_t)NT1 * D_;
  gemm_wmma<1, 0, 0, D_, D_, D_, D_>
      <<<dim3(D_ / 64, NT1 / 128), 256, ldsBytes, stream>>>(
          hln, wqkv_c_h, bqkv_c, qc, nullptr);
  gemm_wmma<1, 0, 0, D_, D_, 2 * D_, D_>
      <<<dim3((2 * D_) / 64, NT2 / 128), 256, ldsBytes, stream>>>(
          pln, wqkv_c_h + (size_t)D_ * D_, bqkv_c + D_, kvc, nullptr);
  flash_attn<D_, 2 * D_, 2 * D_>
      <<<dim3(S1_ / 64, B_ * H_), 128, 0, stream>>>(
          qc, kvc, kvc + D_, attn, S2_, scale);
  gemm_wmma<0, 0, 1, D_, D_, D_, D_>
      <<<dim3(D_ / 64, NT1 / 128), 256, ldsBytes, stream>>>(
          attn, wo_c_h, bo_c, nullptr, x);

  // ---- 3) pre-norm FFN ----------------------------------------------------
  layernorm_f16<<<NT1 / 8, 256, 0, stream>>>(x, g2, be2, hln, NT1);
  gemm_wmma<1, 1, 0, D_, D_, DFF_, D_>
      <<<dim3(DFF_ / 64, NT1 / 128), 256, ldsBytes, stream>>>(
          hln, w1_h, b1, ffnmid, nullptr);
  gemm_wmma<0, 0, 1, DFF_, DFF_, D_, DFF_>
      <<<dim3(D_ / 64, NT1 / 128), 256, ldsBytes, stream>>>(
          ffnmid, w2_h, b2, nullptr, x);
}
